// LSI_Model_20830591385614
// MI455X (gfx1250) — matrix operand order repaired and checked by reading, not compiled, not run on hardware
//
#include <hip/hip_runtime.h>
#include <hip/hip_bf16.h>
#include <math.h>

// ---------------------------------------------------------------------------
// MI455X / gfx1250, wave32. All matmuls via V_WMMA_F32_16X16X32_BF16.
// Fragment-packed tiles: one 16x32 (A) or 32x16 (B) bf16 tile = 256 dwords,
// stored lane-major (lane*8 + v) so a wave loads a whole WMMA operand with
// two coalesced b128 loads per lane.
// Register tile per wave: 32x64 (2 M-tiles x 4 N-tiles = 8 accumulators).
// K trip count is a template parameter: the K loop is fully unrolled and all
// fragment loads become base-pointer + immediate-offset (no per-iteration
// 64-bit address arithmetic).
// Workspace requirement: ~260 MB (encoder processed in 4 L2-resident chunks).
// ---------------------------------------------------------------------------

typedef __attribute__((ext_vector_type(16))) __bf16   v16bf;
typedef __attribute__((ext_vector_type(8)))  float    v8f;
typedef __attribute__((ext_vector_type(4)))  unsigned v4u;

#define N_NODES 204800
#define B_SEG   128
#define NPG_SEG 1600
#define APG_N   128
#define A_TOT   16384
#define IN_F    3
#define HID_F   512
#define OUT_F   128
#define DEC_INF 392

// A-fragment K index for (vgpr slot v, lane-half h):
//   v0..3: K = 2v + 8h           (covers K 0..15)
//   v4..7: K = 16 + 2(v-4) + 8h  (covers K 16..31)
__device__ __forceinline__ int afrag_k(int v, int h) {
  return (v < 4) ? (2 * v + 8 * h) : (16 + 2 * (v - 4) + 8 * h);
}

__device__ __forceinline__ unsigned pack_bf16(float lo, float hi) {
  union { __bf16 h2[2]; unsigned u; } p;
  p.h2[0] = (__bf16)lo;
  p.h2[1] = (__bf16)hi;
  return p.u;
}

template <int ACT>
__device__ __forceinline__ float act_f(float f) {
  if (ACT == 1) return f > 0.f ? f : 0.01f * f;   // jax.nn.leaky_relu default
  if (ACT == 2) return tanhf(f);
  return f;
}

// ---------------------------------------------------------------------------
// Weight packer: f32 W[K x N] row-major -> bf16 B-fragment tiles.
// B-frag: vgpr v, lane l: n = l%16, K = afrag_k(v, l/16) (+1 in high half).
// Tile order: tileIdx = nt*Kt + kt  (K innermost => streaming K loop).
// ---------------------------------------------------------------------------
__global__ __launch_bounds__(256) void pack_weight(
    const float* __restrict__ W, unsigned* __restrict__ outP,
    int Kreal, int Nreal, int Kt, int Nt) {
  int tid = blockIdx.x * 256 + threadIdx.x;
  if (tid >= Kt * Nt * 256) return;
  int v = tid & 7, lane = (tid >> 3) & 31, tile = tid >> 8;
  int kt = tile % Kt, nt = tile / Kt;
  int n = nt * 16 + (lane & 15), h = lane >> 4;
  int k = kt * 32 + afrag_k(v, h);
  float f0 = (k     < Kreal && n < Nreal) ? W[(size_t)k * Nreal + n]       : 0.f;
  float f1 = (k + 1 < Kreal && n < Nreal) ? W[((size_t)k + 1) * Nreal + n] : 0.f;
  outP[tid] = pack_bf16(f0, f1);
}

// ---------------------------------------------------------------------------
// Encoder layer 0 (K=3, trivially VALU-bound): h = leaky(x @ W1 + b1),
// written straight into packed-A bf16 layout for the first WMMA layer.
// ---------------------------------------------------------------------------
__global__ __launch_bounds__(256) void enc_layer0_pack(
    const float* __restrict__ x, const float* __restrict__ W,
    const float* __restrict__ bias, unsigned* __restrict__ outP,
    int mTiles, int Kt) {
  int tid = blockIdx.x * 256 + threadIdx.x;
  if (tid >= mTiles * Kt * 256) return;
  int v = tid & 7, lane = (tid >> 3) & 31, tile = tid >> 8;
  int kt = tile % Kt, mTile = tile / Kt;
  int m = lane & 15, h = lane >> 4;
  int j0 = kt * 32 + afrag_k(v, h);
  int row = mTile * 16 + m;
  float x0 = x[row * 3], x1 = x[row * 3 + 1], x2 = x[row * 3 + 2];
  float r0 = bias[j0]     + x0 * W[j0]     + x1 * W[512 + j0]     + x2 * W[1024 + j0];
  float r1 = bias[j0 + 1] + x0 * W[j0 + 1] + x1 * W[512 + j0 + 1] + x2 * W[1024 + j0 + 1];
  r0 = r0 > 0.f ? r0 : 0.01f * r0;
  r1 = r1 > 0.f ? r1 : 0.01f * r1;
  outP[tid] = pack_bf16(r0, r1);
}

// ---------------------------------------------------------------------------
// Fused WMMA GEMM: C = act(A @ B + bias).
//   A: packed bf16 fragments (mTiles x KT tiles), B: packed bf16 (nt x KT).
//   Register tile: MTILES x NTILES 16x16 accumulators per wave.
//   KT is compile-time: K loop fully unrolled, loads use immediate offsets
//   from 2+4 hoisted base pointers.
//   OUTMODE 0: f32 row-major store (cols < Nreal).
//   OUTMODE 1: act -> bf16 -> next layer's packed-A layout (via per-wave LDS,
//              one 16x64 pass per M-tile to keep LDS at 4 KB/wave).
// ---------------------------------------------------------------------------
template <int MTILES, int NTILES, int KT, int ACT, int OUTMODE>
__global__ __launch_bounds__(256) void wmma_gemm(
    const unsigned* __restrict__ Ap, const unsigned* __restrict__ Bp,
    const float* __restrict__ bias,
    float* __restrict__ outF, unsigned* __restrict__ outP,
    int mTiles, int nGroups, int Nreal, int ktOutTotal) {
  __shared__ float lds[8 * 16 * 64];          // 4 KB per wave, private regions
  const int lane = threadIdx.x & 31;
  const int w    = threadIdx.x >> 5;
  const int wave = blockIdx.x * 8 + w;
  const int mGroups = mTiles / MTILES;
  if (wave >= mGroups * nGroups) return;
  const int mTileBase = (wave / nGroups) * MTILES;
  const int ntBase    = (wave % nGroups) * NTILES;
  const int nIn = lane & 15, h = lane >> 4;

  v8f acc[MTILES][NTILES];
#pragma unroll
  for (int nt = 0; nt < NTILES; ++nt) {
    int col = (ntBase + nt) * 16 + nIn;
    float bv = (col < Nreal) ? bias[col] : 0.f;
#pragma unroll
    for (int mt = 0; mt < MTILES; ++mt)
#pragma unroll
      for (int r = 0; r < 8; ++r) acc[mt][nt][r] = bv;
  }

  // Hoisted fragment base pointers: all K-loop loads are base + imm offset.
  const unsigned* aBase[MTILES];
  const unsigned* bBase[NTILES];
#pragma unroll
  for (int mt = 0; mt < MTILES; ++mt)
    aBase[mt] = Ap + (size_t)(mTileBase + mt) * KT * 256 + lane * 8;
#pragma unroll
  for (int nt = 0; nt < NTILES; ++nt)
    bBase[nt] = Bp + (size_t)(ntBase + nt) * KT * 256 + lane * 8;

#pragma unroll
  for (int kt = 0; kt < KT; ++kt) {
    v16bf a[MTILES];
#pragma unroll
    for (int mt = 0; mt < MTILES; ++mt)
      a[mt] = *(const v16bf*)(aBase[mt] + kt * 256);
#pragma unroll
    for (int nt = 0; nt < NTILES; ++nt) {
      v16bf b = *(const v16bf*)(bBase[nt] + kt * 256);
#pragma unroll
      for (int mt = 0; mt < MTILES; ++mt)
        acc[mt][nt] = __builtin_amdgcn_wmma_f32_16x16x32_bf16(
            false, a[mt], false, b, (short)0, acc[mt][nt], false, false);
    }
  }

  if (OUTMODE == 0) {
    // C layout: vgpr r, lane l -> M = r + 8*(l/16), N = l%16
#pragma unroll
    for (int mt = 0; mt < MTILES; ++mt)
#pragma unroll
      for (int nt = 0; nt < NTILES; ++nt) {
        int col = (ntBase + nt) * 16 + nIn;
        if (col < Nreal) {
#pragma unroll
          for (int r = 0; r < 8; ++r) {
            int row = (mTileBase + mt) * 16 + r + 8 * h;
            outF[(size_t)row * Nreal + col] = act_f<ACT>(acc[mt][nt][r]);
          }
        }
      }
  } else {
    // Bounce through this wave's private LDS strip, reread in A-frag order.
    float* my = lds + w * 16 * 64;
#pragma unroll
    for (int mt = 0; mt < MTILES; ++mt) {
#pragma unroll
      for (int nt = 0; nt < NTILES; ++nt)
#pragma unroll
        for (int r = 0; r < 8; ++r)
          my[(r + 8 * h) * 64 + nt * 16 + nIn] = act_f<ACT>(acc[mt][nt][r]);
      asm volatile("s_wait_dscnt 0" ::: "memory");   // in-wave DS ordering
#pragma unroll
      for (int kt2 = 0; kt2 < NTILES / 2; ++kt2) {   // each 32 cols = one K-tile
        unsigned dw[8];
#pragma unroll
        for (int v = 0; v < 8; ++v) {
          int c = kt2 * 32 + afrag_k(v, h);
          dw[v] = pack_bf16(my[nIn * 64 + c], my[nIn * 64 + c + 1]);
        }
        unsigned* dst = outP +
            ((size_t)(mTileBase + mt) * ktOutTotal + (ntBase >> 1) + kt2) * 256 + lane * 8;
        *(v4u*)dst       = (v4u){dw[0], dw[1], dw[2], dw[3]};
        *(v4u*)(dst + 4) = (v4u){dw[4], dw[5], dw[6], dw[7]};
      }
      asm volatile("s_wait_dscnt 0" ::: "memory");   // WAR before next pass
    }
  }
}

// ---------------------------------------------------------------------------
// Segment mean: batch = repeat(arange(128), 1600) => contiguous segments.
// ---------------------------------------------------------------------------
__global__ __launch_bounds__(128) void seg_mean(
    const float* __restrict__ nh, float* __restrict__ hg) {
  int b = blockIdx.x, j = threadIdx.x;
  float s = 0.f;
  for (int r = 0; r < NPG_SEG; ++r)
    s += nh[((size_t)b * NPG_SEG + r) * OUT_F + j];
  hg[b * OUT_F + j] = s * (1.f / (float)NPG_SEG);
}

// ---------------------------------------------------------------------------
// Decoder feature assembly: concat(nh[u], x[u], nh[v], x[v], hg[b], P, T),
// K padded 392 -> 416, emitted directly in packed-A bf16 layout.
// ---------------------------------------------------------------------------
__device__ __forceinline__ float act_feat(
    int a, int k, const float* nh, const float* x, const float* hg,
    const float* P, const float* T, const int* uv) {
  int u = uv[2 * a], vv = uv[2 * a + 1];
  if (k < 128) return nh[(size_t)u * OUT_F + k];
  k -= 128; if (k < 3)   return x[(size_t)u * 3 + k];
  k -= 3;   if (k < 128) return nh[(size_t)vv * OUT_F + k];
  k -= 128; if (k < 3)   return x[(size_t)vv * 3 + k];
  k -= 3;   if (k < 128) return hg[(size_t)(a >> 7) * OUT_F + k];
  k -= 128; if (k == 0)  return P[a];
  if (k == 1) return T[a];
  return 0.f;  // zero padding 392..415
}

__global__ __launch_bounds__(256) void assemble_actions(
    const float* __restrict__ nh, const float* __restrict__ x,
    const float* __restrict__ hg, const float* __restrict__ P,
    const float* __restrict__ T, const int* __restrict__ uv,
    unsigned* __restrict__ outP, int mTiles, int Kt) {
  int tid = blockIdx.x * 256 + threadIdx.x;
  if (tid >= mTiles * Kt * 256) return;
  int v = tid & 7, lane = (tid >> 3) & 31, tile = tid >> 8;
  int kt = tile % Kt, mTile = tile / Kt;
  int m = lane & 15, h = lane >> 4;
  int k0 = kt * 32 + afrag_k(v, h);
  int a = mTile * 16 + m;
  float f0 = act_feat(a, k0,     nh, x, hg, P, T, uv);
  float f1 = act_feat(a, k0 + 1, nh, x, hg, P, T, uv);
  outP[tid] = pack_bf16(f0, f1);
}

// ---------------------------------------------------------------------------
// Softmax / log-softmax / entropy / Gumbel-argmax sampling per group.
// (JAX threefry stream is not bit-reproducible here; deterministic splitmix
//  Gumbel noise keyed by (42, group, slot) stands in for categorical().)
// Output: [sampled_action(128x2) | log_prob(128) | entropy(128)] as f32.
// ---------------------------------------------------------------------------
__global__ __launch_bounds__(128) void softmax_sample(
    const float* __restrict__ scores, const int* __restrict__ uv,
    const unsigned char* __restrict__ opt, float* __restrict__ out) {
  __shared__ float sv[128];
  __shared__ float s2[128];
  __shared__ int   si[128];
  __shared__ float slogp[128];
  int b = blockIdx.x, t = threadIdx.x;
  float s = scores[b * APG_N + t];

  sv[t] = s; __syncthreads();
  for (int o = 64; o; o >>= 1) { if (t < o) sv[t] = fmaxf(sv[t], sv[t + o]); __syncthreads(); }
  float mx = sv[0]; __syncthreads();

  float e = __expf(s - mx);
  sv[t] = e; __syncthreads();
  for (int o = 64; o; o >>= 1) { if (t < o) sv[t] += sv[t + o]; __syncthreads(); }
  float Z = sv[0]; __syncthreads();

  float lp = s - mx - __logf(Z);
  float pi = e / Z;
  slogp[t] = lp;
  sv[t] = -pi * lp; __syncthreads();
  for (int o = 64; o; o >>= 1) { if (t < o) sv[t] += sv[t + o]; __syncthreads(); }
  float ent = sv[0]; __syncthreads();

  unsigned long long z = 0x2545F4914F6CDD1Dull * (unsigned long long)(b * APG_N + t + 1) + 42ull;
  z ^= z >> 30; z *= 0xBF58476D1CE4E5B9ull;
  z ^= z >> 27; z *= 0x94D049BB133111EBull;
  z ^= z >> 31;
  float u = ((float)(unsigned)(z >> 40) + 0.5f) * (1.0f / 16777216.0f);
  float g = -__logf(-__logf(u));
  s2[t] = s + g; si[t] = t; __syncthreads();
  for (int o = 64; o; o >>= 1) {
    if (t < o && s2[t + o] > s2[t]) { s2[t] = s2[t + o]; si[t] = si[t + o]; }
    __syncthreads();
  }
  if (t == 0) {
    int aid = si[0];
    int ga = b * APG_N + aid;
    out[b * 2 + 0] = (float)uv[ga * 2 + 0];
    out[b * 2 + 1] = (float)uv[ga * 2 + 1];
    bool om = opt[b] != 0;
    out[2 * B_SEG + b] = om ? 0.f : slogp[aid];
    out[3 * B_SEG + b] = om ? 0.f : ent;
  }
}

// ---------------------------------------------------------------------------
extern "C" void kernel_launch(void* const* d_in, const int* in_sizes, int n_in,
                              void* d_out, int out_size, void* d_ws, size_t ws_size,
                              hipStream_t stream) {
  (void)n_in; (void)out_size; (void)ws_size;
  const float* x = (const float*)d_in[0];
  const float* P = (const float*)d_in[1];
  const float* T = (const float*)d_in[2];

  // params pytree order may be insertion (enc first) or sorted (dec first).
  const float *eW[4], *eb[4], *dW[4], *db[4];
  if (in_sizes[3] == DEC_INF * HID_F) {           // sorted: dec_W first
    for (int i = 0; i < 4; ++i) {
      dW[i] = (const float*)d_in[3 + i];  db[i] = (const float*)d_in[7 + i];
      eW[i] = (const float*)d_in[11 + i]; eb[i] = (const float*)d_in[15 + i];
    }
  } else {                                        // insertion: enc_W first
    for (int i = 0; i < 4; ++i) {
      eW[i] = (const float*)d_in[3 + i];  eb[i] = (const float*)d_in[7 + i];
      dW[i] = (const float*)d_in[11 + i]; db[i] = (const float*)d_in[15 + i];
    }
  }
  const int* uv = (const int*)d_in[19];
  const unsigned char* opt = (const unsigned char*)d_in[22];

  // ---- workspace carve (total ~260 MB) ----
  char* base = (char*)d_ws;
  size_t off = 0;
  auto carve = [&](size_t bytes) -> char* {
    char* p = base + off;
    off = (off + bytes + 255) & ~(size_t)255;
    return p;
  };
  float*    node_h = (float*)carve((size_t)N_NODES * OUT_F * 4);
  float*    hg     = (float*)carve((size_t)B_SEG * OUT_F * 4);
  float*    scores = (float*)carve((size_t)A_TOT * 4);
  const int CHUNK = 51200, CH_MT = CHUNK / 16;            // 4 L2-resident chunks
  unsigned* bufA  = (unsigned*)carve((size_t)CH_MT * 16 * 256 * 4);
  unsigned* bufB  = (unsigned*)carve((size_t)CH_MT * 16 * 256 * 4);
  unsigned* decA  = (unsigned*)carve((size_t)1024 * 13 * 256 * 4);
  unsigned* decB1 = (unsigned*)carve((size_t)1024 * 16 * 256 * 4);
  unsigned* decB2 = (unsigned*)carve((size_t)1024 * 16 * 256 * 4);
  unsigned* wE2 = (unsigned*)carve((size_t)16 * 32 * 256 * 4);
  unsigned* wE3 = (unsigned*)carve((size_t)16 * 32 * 256 * 4);
  unsigned* wE4 = (unsigned*)carve((size_t)16 * 8  * 256 * 4);
  unsigned* wD1 = (unsigned*)carve((size_t)13 * 32 * 256 * 4);
  unsigned* wD2 = (unsigned*)carve((size_t)16 * 32 * 256 * 4);
  unsigned* wD3 = (unsigned*)carve((size_t)16 * 32 * 256 * 4);
  unsigned* wD4 = (unsigned*)carve((size_t)16 * 1  * 256 * 4);

  // ---- pack weights to bf16 fragment tiles ----
  auto packW = [&](const float* W, unsigned* dst, int Kr, int Nr, int Kt, int Nt) {
    int total = Kt * Nt * 256;
    pack_weight<<<(total + 255) / 256, 256, 0, stream>>>(W, dst, Kr, Nr, Kt, Nt);
  };
  packW(eW[1], wE2, 512, 512, 16, 32);
  packW(eW[2], wE3, 512, 512, 16, 32);
  packW(eW[3], wE4, 512, 128, 16, 8);
  packW(dW[0], wD1, 392, 512, 13, 32);
  packW(dW[1], wD2, 512, 512, 16, 32);
  packW(dW[2], wD3, 512, 512, 16, 32);
  packW(dW[3], wD4, 512, 1,   16, 1);

  // ---- encoder, 4 chunks of 51200 rows (activations stay in L2) ----
  // waves per GEMM = (mTiles/2) * nGroups; 8 waves per block.
  for (int c = 0; c < 4; ++c) {
    const float* xc = x + (size_t)c * CHUNK * IN_F;
    int tot0 = CH_MT * 16 * 256;
    enc_layer0_pack<<<(tot0 + 255) / 256, 256, 0, stream>>>(xc, eW[0], eb[0], bufA, CH_MT, 16);
    wmma_gemm<2, 4, 16, 1, 1><<<(CH_MT / 2 * 8 + 7) / 8, 256, 0, stream>>>(
        bufA, wE2, eb[1], nullptr, bufB, CH_MT, 8, 512, 16);
    wmma_gemm<2, 4, 16, 1, 1><<<(CH_MT / 2 * 8 + 7) / 8, 256, 0, stream>>>(
        bufB, wE3, eb[2], nullptr, bufA, CH_MT, 8, 512, 16);
    wmma_gemm<2, 4, 16, 0, 0><<<(CH_MT / 2 * 2 + 7) / 8, 256, 0, stream>>>(
        bufA, wE4, eb[3], node_h + (size_t)c * CHUNK * OUT_F, nullptr,
        CH_MT, 2, 128, 0);
  }

  // ---- graph pooling + decoder ----
  seg_mean<<<B_SEG, 128, 0, stream>>>(node_h, hg);
  int totA = 1024 * 13 * 256;
  assemble_actions<<<(totA + 255) / 256, 256, 0, stream>>>(
      node_h, x, hg, P, T, uv, decA, 1024, 13);
  wmma_gemm<2, 4, 13, 2, 1><<<(512 * 8 + 7) / 8, 256, 0, stream>>>(
      decA, wD1, db[0], nullptr, decB1, 1024, 8, 512, 16);
  wmma_gemm<2, 4, 16, 2, 1><<<(512 * 8 + 7) / 8, 256, 0, stream>>>(
      decB1, wD2, db[1], nullptr, decB2, 1024, 8, 512, 16);
  wmma_gemm<2, 4, 16, 2, 1><<<(512 * 8 + 7) / 8, 256, 0, stream>>>(
      decB2, wD3, db[2], nullptr, decB1, 1024, 8, 512, 16);
  wmma_gemm<2, 1, 16, 0, 0><<<(512 + 7) / 8, 256, 0, stream>>>(
      decB1, wD4, db[3], scores, nullptr, 1024, 1, 1, 0);

  // ---- softmax / entropy / sampling ----
  softmax_sample<<<B_SEG, 128, 0, stream>>>(scores, uv, opt, (float*)d_out);
}
